// SetAbstractionMSG_54219667144941
// MI455X (gfx1250) — matrix operand order repaired and checked by reading, not compiled, not run on hardware
//
#include <hip/hip_runtime.h>

// ---------------------------------------------------------------------------
// PointNet++ MSG Set-Abstraction on CDNA5 (gfx1250, wave32, WMMA)
//   B=4, N=16384, S=2048, C_IN=16, radii (0.4,0.8], K=(16,32),
//   MLPs (19->32->32->64) x2 scales, agg 128->64, training-mode BN everywhere.
// ---------------------------------------------------------------------------

#define Bb   4
#define Nn   16384
#define Ss   2048
#define CIN  16
#define AGG  64
#define EPSB 1e-5f

typedef __attribute__((ext_vector_type(16))) _Float16 v16h;
typedef __attribute__((ext_vector_type(8)))  _Float16 v8h;
typedef __attribute__((ext_vector_type(8)))  float    v8f;

// ---------------------------------------------------------------------------
// 1) Farthest point sampling: one 1024-thread block per batch.
//    Each thread owns 16 points in registers; per step: update min-dist,
//    wave32 shuffle argmax, cross-wave LDS argmax, broadcast winner.
// ---------------------------------------------------------------------------
__global__ __launch_bounds__(1024)
void fps_kernel(const float* __restrict__ xyz, int* __restrict__ idx) {
    const int b = blockIdx.x, t = threadIdx.x;
    const float* base = xyz + (size_t)b * Nn * 3;
    float px[16], py[16], pz[16], mind[16];
#pragma unroll
    for (int i = 0; i < 16; ++i) {
        int p = i * 1024 + t;
        px[i] = base[p * 3 + 0];
        py[i] = base[p * 3 + 1];
        pz[i] = base[p * 3 + 2];
        mind[i] = 1e10f;
    }
    if (t == 0) idx[(size_t)b * Ss] = 0;

    __shared__ float sval[32];
    __shared__ int   sidx[32];
    __shared__ int   snext;

    const int lane = t & 31, wv = t >> 5;
    int last = 0;
    for (int s = 1; s < Ss; ++s) {
        float lx = base[last * 3 + 0];
        float ly = base[last * 3 + 1];
        float lz = base[last * 3 + 2];
        float best = -1.0f; int bi = 0;
#pragma unroll
        for (int i = 0; i < 16; ++i) {
            float dx = px[i] - lx, dy = py[i] - ly, dz = pz[i] - lz;
            float d = dx * dx + dy * dy + dz * dz;
            mind[i] = fminf(mind[i], d);
            if (mind[i] > best) { best = mind[i]; bi = i * 1024 + t; }
        }
#pragma unroll
        for (int off = 16; off >= 1; off >>= 1) {
            float ov = __shfl_xor(best, off, 32);
            int   oi = __shfl_xor(bi,   off, 32);
            if (ov > best || (ov == best && oi < bi)) { best = ov; bi = oi; }
        }
        if (lane == 0) { sval[wv] = best; sidx[wv] = bi; }
        __syncthreads();
        if (wv == 0) {
            best = sval[lane]; bi = sidx[lane];
#pragma unroll
            for (int off = 16; off >= 1; off >>= 1) {
                float ov = __shfl_xor(best, off, 32);
                int   oi = __shfl_xor(bi,   off, 32);
                if (ov > best || (ov == best && oi < bi)) { best = ov; bi = oi; }
            }
            if (lane == 0) { snext = bi; idx[(size_t)b * Ss + s] = bi; }
        }
        __syncthreads();
        last = snext;
    }
}

// ---------------------------------------------------------------------------
// 2) Gather sampled centers -> d_out[0 : B*S*3]
// ---------------------------------------------------------------------------
__global__ void gather_centers(const float* __restrict__ xyz,
                               const int* __restrict__ idx,
                               float* __restrict__ newxyz) {
    int t = blockIdx.x * blockDim.x + threadIdx.x;
    if (t >= Bb * Ss) return;
    int b = t >> 11;
    int i = idx[t];
    const float* q = xyz + ((size_t)b * Nn + i) * 3;
    newxyz[t * 3 + 0] = q[0];
    newxyz[t * 3 + 1] = q[1];
    newxyz[t * 3 + 2] = q[2];
}

// ---------------------------------------------------------------------------
// 3) Dilated ball query: one wave per center, ballot-compaction keeps the
//    first-K-by-index semantics of the reference top_k trick. Early exit is
//    wave-uniform (counts derive from ballots).
// ---------------------------------------------------------------------------
__global__ void ball_query(const float* __restrict__ xyz,
                           const float* __restrict__ newxyz,
                           int* __restrict__ nbr1, int* __restrict__ nbr2) {
    const int lane = threadIdx.x & 31;
    const int wv = (blockIdx.x * blockDim.x + threadIdx.x) >> 5;
    if (wv >= Bb * Ss) return;
    const int b = wv >> 11;
    const float cx = newxyz[wv * 3 + 0];
    const float cy = newxyz[wv * 3 + 1];
    const float cz = newxyz[wv * 3 + 2];
    const float* base = xyz + (size_t)b * Nn * 3;
    int* row1 = nbr1 + (size_t)wv * 16;
    int* row2 = nbr2 + (size_t)wv * 32;
    const float r1sq = 0.4f * 0.4f, r2sq = 0.8f * 0.8f;
    const unsigned lt = (1u << lane) - 1u;
    int c1 = 0, c2 = 0;
    for (int bj = 0; bj < Nn; bj += 32) {
        int j = bj + lane;
        float dx = base[j * 3 + 0] - cx;
        float dy = base[j * 3 + 1] - cy;
        float dz = base[j * 3 + 2] - cz;
        float d2 = dx * dx + dy * dy + dz * dz;
        bool in1 = d2 < r1sq;
        bool in2 = (d2 < r2sq) && (d2 >= r1sq);
        unsigned m1 = (unsigned)__ballot(in1);
        unsigned m2 = (unsigned)__ballot(in2);
        if (in1) { int r = c1 + __popc(m1 & lt); if (r < 16) row1[r] = j; }
        if (in2) { int r = c2 + __popc(m2 & lt); if (r < 32) row2[r] = j; }
        c1 = min(16, c1 + __popc(m1));
        c2 = min(32, c2 + __popc(m2));
        if (c1 >= 16 && c2 >= 32) break;
    }
    // Pad with first hit; empty ball -> index 0 (matches reference).
    int f1 = (c1 > 0) ? row1[0] : 0;
    int f2 = (c2 > 0) ? row2[0] : 0;
    if (lane >= c1 && lane < 16) row1[lane] = f1;
    if (lane >= c2)              row2[lane] = f2;
}

// ---------------------------------------------------------------------------
// 4) Build f16 operand rows: [dx,dy,dz, fea(16), zero-pad to 32]
//    Row assembled in registers, written with 4x 16B vector stores.
// ---------------------------------------------------------------------------
__global__ void gather_ball(const float* __restrict__ xyz,
                            const float* __restrict__ fea,
                            const float* __restrict__ newxyz,
                            const int* __restrict__ nbr,
                            _Float16* __restrict__ Ah, int K, int Mtot) {
    int r = blockIdx.x * blockDim.x + threadIdx.x;
    if (r >= Mtot) return;
    int p = r / K;         // b*S + s
    int b = p >> 11;
    int j = nbr[r];
    const float* c3 = newxyz + (size_t)p * 3;
    const float* q  = xyz + ((size_t)b * Nn + j) * 3;
    _Float16 buf[32];
    buf[0] = (_Float16)(q[0] - c3[0]);
    buf[1] = (_Float16)(q[1] - c3[1]);
    buf[2] = (_Float16)(q[2] - c3[2]);
#pragma unroll
    for (int c = 0; c < CIN; ++c)
        buf[3 + c] = (_Float16)fea[((size_t)b * CIN + c) * Nn + j];
#pragma unroll
    for (int c = 3 + CIN; c < 32; ++c) buf[c] = (_Float16)0.0f;
    v8h* dst = (v8h*)(Ah + (size_t)r * 32);
#pragma unroll
    for (int qd = 0; qd < 4; ++qd) {
        v8h x;
#pragma unroll
        for (int e = 0; e < 8; ++e) x[e] = buf[qd * 8 + e];
        dst[qd] = x;
    }
}

// ---------------------------------------------------------------------------
// 5) Convert weights (Cin x Cout f32) directly into the contiguous-k B-fragment order mm_wmma uses:
//    Wf[((kb*TN + tn)*32 + lane)*16 + e] = W[kb*32 + 16*(lane>>4) + e][tn*16 + (lane&15)]
//    so the GEMM fetches each lane's fragment with contiguous 16B loads.
// ---------------------------------------------------------------------------
__global__ void cvt_w_frag(const float* __restrict__ W, _Float16* __restrict__ Wf,
                           int Cin, int Cout, int Kp) {
    int t = blockIdx.x * blockDim.x + threadIdx.x;
    if (t >= Kp * Cout) return;
    int e    = t & 15;
    int l    = (t >> 4) & 31;
    int rest = t >> 9;                 // kb*TN + tn
    int TN   = Cout >> 4;
    int kb = rest / TN, tn = rest % TN;
    int n  = tn * 16 + (l & 15);
    int k  = kb * 32 + 16 * (l >> 4) + e;
    Wf[t] = (k < Cin) ? (_Float16)W[k * Cout + n] : (_Float16)0.0f;
}

// ---------------------------------------------------------------------------
// 6) WMMA GEMM:  Z[M x TN*16] = Ah[M x Kp] * W + bias
//    One wave per 16-row strip; A fragment loaded once per k-step and reused
//    across all TN column tiles (TN unrolled v_wmma per step). Cout is a
//    compile-time constant so store addressing strength-reduces.
// ---------------------------------------------------------------------------
template <int TN>
__global__ void mm_wmma(const _Float16* __restrict__ A,
                        const _Float16* __restrict__ Wf,
                        const float* __restrict__ bias,
                        float* __restrict__ Z, int M, int Kp) {
    constexpr int Cout = TN * 16;
    int gtid = blockIdx.x * blockDim.x + threadIdx.x;
    int w = gtid >> 5;
    if (w >= (M >> 4)) return;
    int lane = threadIdx.x & 31;
    int m = lane & 15, hi = lane >> 4;

    const _Float16* arow = A + (size_t)(w * 16 + m) * Kp;
    const v16h* bbase = (const v16h*)Wf;
    v8f acc[TN];
#pragma unroll
    for (int tn = 0; tn < TN; ++tn) acc[tn] = (v8f){};

    for (int kk = 0, kb = 0; kk < Kp; kk += 32, ++kb) {
        // A 16x32 f16 fragment: lanes 0-15 carry K=0..15, lanes 16-31 K=16..31
        v8h alo = *(const v8h*)(arow + kk + 16 * hi);
        v8h ahi = *(const v8h*)(arow + kk + 16 * hi + 8);
        v16h a;
#pragma unroll
        for (int e = 0; e < 8; ++e) { a[e] = alo[e]; a[8 + e] = ahi[e]; }
#pragma unroll
        for (int tn = 0; tn < TN; ++tn) {
            v16h bf = bbase[(size_t)(kb * TN + tn) * 32 + lane];
            acc[tn] = __builtin_amdgcn_wmma_f32_16x16x32_f16(false, a, false, bf,
                                                             (short)0, acc[tn], false, false);
        }
    }
#pragma unroll
    for (int tn = 0; tn < TN; ++tn) {
        float bv = bias[tn * 16 + m];
#pragma unroll
        for (int v = 0; v < 8; ++v) {
            int row = w * 16 + v + 8 * hi;        // D layout: VGPR v -> M = v (+8 for hi lanes)
            Z[(size_t)row * Cout + tn * 16 + m] = acc[tn][v] + bv;
        }
    }
}

// ---------------------------------------------------------------------------
// 7) Per-channel sum / sum-of-squares over M rows (for training-mode BN)
// ---------------------------------------------------------------------------
__global__ void stats_kernel(const float* __restrict__ Z, int M, int C,
                             float* __restrict__ stats) {
    __shared__ float ssum[256], ssq[256];
    int t = threadIdx.x;
    int c = t % C;
    int rgrp = t / C;
    int rpb = blockDim.x / C;
    float s = 0.f, q = 0.f;
    for (size_t r = (size_t)blockIdx.x * rpb + rgrp; r < (size_t)M;
         r += (size_t)gridDim.x * rpb) {
        float v = Z[r * C + c];
        s += v; q += v * v;
    }
    ssum[t] = s; ssq[t] = q;
    __syncthreads();
    for (int off = blockDim.x / 2; off >= C; off >>= 1) {
        if (t < off) { ssum[t] += ssum[t + off]; ssq[t] += ssq[t + off]; }
        __syncthreads();
    }
    if (t < C) {
        atomicAdd(&stats[c], ssum[t]);
        atomicAdd(&stats[64 + c], ssq[t]);
    }
}

__global__ void bn_finalize(float* __restrict__ stats,
                            const float* __restrict__ gamma, int C, float invM) {
    int c = threadIdx.x;
    if (c >= C) return;
    float mean = stats[c] * invM;
    float var  = stats[64 + c] * invM - mean * mean;   // biased var (jnp.var)
    stats[128 + c] = mean;
    stats[192 + c] = gamma[c] * rsqrtf(var + EPSB);
}

// 8) BN + ReLU -> f16 operand for next layer (zero-padded to Kp columns)
__global__ void bn_relu_to_h(const float* __restrict__ Z,
                             const float* __restrict__ stats,
                             const float* __restrict__ beta,
                             _Float16* __restrict__ Ah, int C, int Kp, int total) {
    int t = blockIdx.x * blockDim.x + threadIdx.x;
    if (t >= total) return;
    int c = t % Kp;
    size_t r = (size_t)(t / Kp);
    float v = 0.f;
    if (c < C) {
        float z = Z[r * C + c];
        v = fmaxf(stats[192 + c] * (z - stats[128 + c]) + beta[c], 0.f);
    }
    Ah[t] = (_Float16)v;
}

// 9) BN + ReLU + max over the K neighbors -> pooled (B*S, 64)
__global__ void bn_relu_max(const float* __restrict__ Z,
                            const float* __restrict__ stats,
                            const float* __restrict__ beta,
                            float* __restrict__ outp, int K, int total) {
    int t = blockIdx.x * blockDim.x + threadIdx.x;
    if (t >= total) return;
    int c = t % 64, g = t / 64;
    float sc = stats[192 + c], mu = stats[128 + c], be = beta[c];
    float m = 0.f;   // ReLU outputs are >= 0
    for (int k = 0; k < K; ++k) {
        float z = Z[((size_t)g * K + k) * 64 + c];
        m = fmaxf(m, fmaxf(sc * (z - mu) + be, 0.f));
    }
    outp[t] = m;
}

// 10) Concatenate the two pooled scales -> f16 (B*S, 128)
__global__ void gather_agg(const float* __restrict__ o1,
                           const float* __restrict__ o2,
                           _Float16* __restrict__ Ah, int total) {
    int t = blockIdx.x * blockDim.x + threadIdx.x;
    if (t >= total) return;
    int c = t % 128, p = t / 128;
    float v = (c < 64) ? o1[(size_t)p * 64 + c] : o2[(size_t)p * 64 + (c - 64)];
    Ah[t] = (_Float16)v;
}

// 11) Final BN + ReLU, written transposed as (B, 64, S)
__global__ void bn_relu_transpose(const float* __restrict__ Z,
                                  const float* __restrict__ stats,
                                  const float* __restrict__ beta,
                                  float* __restrict__ outp, int total) {
    int t = blockIdx.x * blockDim.x + threadIdx.x;
    if (t >= total) return;
    int s = t % Ss;
    int bc = t / Ss;
    int c = bc % 64, b = bc / 64;
    float z = Z[((size_t)(b * Ss + s)) * 64 + c];
    outp[t] = fmaxf(stats[192 + c] * (z - stats[128 + c]) + beta[c], 0.f);
}

// ---------------------------------------------------------------------------
// Host-side orchestration
// ---------------------------------------------------------------------------
extern "C" void kernel_launch(void* const* d_in, const int* in_sizes, int n_in,
                              void* d_out, int out_size, void* d_ws, size_t ws_size,
                              hipStream_t stream) {
    (void)in_sizes; (void)n_in; (void)out_size; (void)ws_size;

    const float* xyz = (const float*)d_in[0];   // (B,N,3)
    const float* fea = (const float*)d_in[1];   // (B,C_IN,N)
    // mlp_params[s][l] = (W, b, gamma, beta) flattened: scale0 at 2.., scale1 at 14.., agg at 26..
    auto P = [&](int i) { return (const float*)d_in[i]; };

    float* out_xyz = (float*)d_out;                       // (B,S,3)
    float* out_fea = (float*)d_out + (size_t)Bb * Ss * 3; // (B,64,S)

    // ---- workspace carve-out (all offsets 256B aligned) ----
    char* ws = (char*)d_ws;
    size_t off = 0;
    auto carve = [&](size_t bytes) -> void* {
        off = (off + 255) & ~(size_t)255;
        void* p = ws + off;
        off += bytes;
        return p;
    };
    const int M1 = Bb * Ss * 16;      // 131072
    const int M2 = Bb * Ss * 32;      // 262144
    const int Mg = Bb * Ss;           // 8192

    int*      w_idx  = (int*)carve((size_t)Bb * Ss * 4);
    int*      w_nbr1 = (int*)carve((size_t)M1 * 4);
    int*      w_nbr2 = (int*)carve((size_t)M2 * 4);
    float*    w_st   = (float*)carve(256 * 4);               // sum|sumsq|mean|scale
    _Float16* w_Wh[7];
    const int whalfs[7] = {32 * 32, 32 * 32, 32 * 64, 32 * 32, 32 * 32, 32 * 64, 128 * 64};
    for (int i = 0; i < 7; ++i) w_Wh[i] = (_Float16*)carve((size_t)whalfs[i] * 2);
    float*    w_out1 = (float*)carve((size_t)Mg * 64 * 4);
    float*    w_out2 = (float*)carve((size_t)Mg * 64 * 4);
    _Float16* w_Ah   = (_Float16*)carve((size_t)M2 * 32 * 2); // also holds 8192x128 agg operand
    float*    w_Z    = (float*)carve((size_t)M2 * 64 * 4);

    const int T = 256;
    auto blk = [&](long n) { return (int)((n + T - 1) / T); };

    // ---- sampling / grouping ----
    fps_kernel<<<Bb, 1024, 0, stream>>>(xyz, w_idx);
    gather_centers<<<blk(Bb * Ss), T, 0, stream>>>(xyz, w_idx, out_xyz);
    ball_query<<<blk((long)Bb * Ss * 32), T, 0, stream>>>(xyz, out_xyz, w_nbr1, w_nbr2);

    // ---- weight conversion (f32 -> B-fragment-ordered f16) ----
    const int wCin[7]  = {19, 32, 32, 19, 32, 32, 128};
    const int wCout[7] = {32, 32, 64, 32, 32, 64, 64};
    const int wKp[7]   = {32, 32, 32, 32, 32, 32, 128};
    const int wIn[7]   = {2, 6, 10, 14, 18, 22, 26};
    for (int i = 0; i < 7; ++i)
        cvt_w_frag<<<blk((long)wKp[i] * wCout[i]), T, 0, stream>>>(
            P(wIn[i]), w_Wh[i], wCin[i], wCout[i], wKp[i]);

    auto mm = [&](const _Float16* A, const _Float16* W, const float* bias,
                  float* Z, int M, int Kp, int C) {
        long waves = (long)(M / 16);
        if (C == 32)
            mm_wmma<2><<<blk(waves * 32), T, 0, stream>>>(A, W, bias, Z, M, Kp);
        else
            mm_wmma<4><<<blk(waves * 32), T, 0, stream>>>(A, W, bias, Z, M, Kp);
    };
    auto bn_pass = [&](float* Z, int M, int C, const float* gamma) {
        hipMemsetAsync(w_st, 0, 128 * sizeof(float), stream);
        stats_kernel<<<256, T, 0, stream>>>(Z, M, C, w_st);
        bn_finalize<<<1, 64, 0, stream>>>(w_st, gamma, C, 1.0f / (float)M);
    };

    // ---- two grouping scales: MLP 19->32->32->64 with BN, max over K ----
    const int Ms[2] = {M1, M2};
    const int Ks[2] = {16, 32};
    int* nbrs[2] = {w_nbr1, w_nbr2};
    float* pooled[2] = {w_out1, w_out2};
    for (int sidx = 0; sidx < 2; ++sidx) {
        const int M = Ms[sidx], K = Ks[sidx], base = 2 + 12 * sidx;
        gather_ball<<<blk(M), T, 0, stream>>>(xyz, fea, out_xyz, nbrs[sidx], w_Ah, K, M);
        // layer 1: 19(pad32) -> 32
        mm(w_Ah, w_Wh[3 * sidx + 0], P(base + 1), w_Z, M, 32, 32);
        bn_pass(w_Z, M, 32, P(base + 2));
        bn_relu_to_h<<<blk((long)M * 32), T, 0, stream>>>(w_Z, w_st, P(base + 3), w_Ah, 32, 32, M * 32);
        // layer 2: 32 -> 32
        mm(w_Ah, w_Wh[3 * sidx + 1], P(base + 5), w_Z, M, 32, 32);
        bn_pass(w_Z, M, 32, P(base + 6));
        bn_relu_to_h<<<blk((long)M * 32), T, 0, stream>>>(w_Z, w_st, P(base + 7), w_Ah, 32, 32, M * 32);
        // layer 3: 32 -> 64, then BN+ReLU+max over K
        mm(w_Ah, w_Wh[3 * sidx + 2], P(base + 9), w_Z, M, 32, 64);
        bn_pass(w_Z, M, 64, P(base + 10));
        bn_relu_max<<<blk((long)Mg * 64), T, 0, stream>>>(w_Z, w_st, P(base + 11),
                                                          pooled[sidx], K, Mg * 64);
    }

    // ---- aggregation MLP 128 -> 64, BN+ReLU, transpose to (B,64,S) ----
    gather_agg<<<blk((long)Mg * 128), T, 0, stream>>>(w_out1, w_out2, w_Ah, Mg * 128);
    mm(w_Ah, w_Wh[6], P(27), w_Z, Mg, 128, 64);
    bn_pass(w_Z, Mg, 64, P(28));
    bn_relu_transpose<<<blk((long)Bb * 64 * Ss), T, 0, stream>>>(w_Z, w_st, P(29),
                                                                 out_fea, Bb * 64 * Ss);
}
